// SoftmaxAttention_37460704756493
// MI455X (gfx1250) — compile-verified, matrix operand order checked by reading, not run on hardware
//
#include <hip/hip_runtime.h>
#include <hip/hip_bf16.h>

// Problem constants (match reference): B=4, T=2048, D=512, M=8 heads.
static constexpr int B_ = 4;
static constexpr int T_ = 2048;
static constexpr int D_ = 512;
static constexpr int M_ = 8;
static constexpr float LN_EPS_ = 1e-5f;

typedef __attribute__((ext_vector_type(16))) _Float16 v16h;
typedef __attribute__((ext_vector_type(8)))  _Float16 v8h;
typedef __attribute__((ext_vector_type(8)))  float    v8f;
typedef __attribute__((ext_vector_type(4)))  unsigned int v4u;
typedef __attribute__((ext_vector_type(4)))  int v4i;
typedef __attribute__((ext_vector_type(8)))  int v8i;

// WMMA D = A*B + C, f16 inputs, f32 accum (16x16x32).
__device__ __forceinline__ v8f wmma16(v16h a, v16h b, v8f c) {
  return __builtin_amdgcn_wmma_f32_16x16x32_f16(
      /*neg_a=*/false, a, /*neg_b=*/false, b,
      /*c_mod=*/(short)0, c, /*reuse_a=*/false, /*reuse_b=*/false);
}

// Load a 16x32 (16-bit) WMMA fragment from an LDS tile.
// Works for A (rows = M, contiguous K) and B stored as [N][K] rows.
// ISA layout: lanes 0-15 hold K={0..7,16..23}, lanes 16-31 hold K={8..15,24..31},
// each half as two contiguous 8-element (16B) chunks.
__device__ __forceinline__ v16h ld_frag(const _Float16* lds, int stride,
                                        int row0, int k0, int lane) {
  const _Float16* p = lds + (size_t)(row0 + (lane & 15)) * stride + k0 + ((lane >> 4) << 3);
  v8h lo = *(const v8h*)p;
  v8h hi = *(const v8h*)(p + 16);
  v16h r;
#pragma unroll
  for (int i = 0; i < 8; ++i) { r[i] = lo[i]; r[i + 8] = hi[i]; }
  return r;
}

// ---------------------------------------------------------------------------
// Tensor Data Mover: 2D tile load (16-bit elements) global -> LDS.
// D# bit layout per CDNA5 ISA.
// tile_d0 = contiguous elements per row, tile_d1 = rows,
// stride0 = row pitch in elements, tensor_d0/d1 = remaining extent for OOB.
// Must be executed by ONE wave per workgroup (TDM ignores EXEC).
// ---------------------------------------------------------------------------
__device__ __forceinline__ void tdm_load_2d(const void* gptr, unsigned lds_off,
                                            unsigned tile_d0, unsigned tile_d1,
                                            unsigned tensor_d0, unsigned tensor_d1,
                                            unsigned long long stride0) {
  unsigned long long ga = (unsigned long long)(uintptr_t)gptr;
  v4u g0;
  g0[0] = 1u;                                           // count=1 (valid), user mode
  g0[1] = lds_off;                                      // lds_addr [63:32]
  g0[2] = (unsigned)ga;                                 // global_addr[31:0]
  g0[3] = (unsigned)((ga >> 32) & 0x1FFFFFFu) | (2u << 30); // addr[56:32] | type=2
  v8i g1;
  g1[0] = (int)(1u << 16);                              // wg_mask=0 | data_size=1 (2B)
  g1[1] = (int)((tensor_d0 & 0xFFFFu) << 16);           // bar_addr=0 | tensor_dim0 lo
  g1[2] = (int)(((tensor_d0 >> 16) & 0xFFFFu) | ((tensor_d1 & 0xFFFFu) << 16));
  g1[3] = (int)(((tensor_d1 >> 16) & 0xFFFFu) | ((tile_d0 & 0xFFFFu) << 16));
  g1[4] = (int)(tile_d1 & 0xFFFFu);                     // tile_dim1 | tile_dim2=0
  g1[5] = (int)(unsigned)(stride0 & 0xFFFFFFFFu);       // dim0_stride lo32
  g1[6] = (int)((unsigned)(stride0 >> 32) & 0xFFFFu);   // dim0_stride hi16 | dim1_stride lo=0
  g1[7] = 0;
  v4i z4 = {0, 0, 0, 0};
#if __clang_major__ >= 23
  v8i z8 = {0, 0, 0, 0, 0, 0, 0, 0};
  __builtin_amdgcn_tensor_load_to_lds(g0, g1, z4, z4, z8, 0);
#else
  __builtin_amdgcn_tensor_load_to_lds(g0, g1, z4, z4, 0);
#endif
}

__device__ __forceinline__ unsigned lds_offset_of(const void* p) {
  return (unsigned)(uintptr_t)p;   // flat LDS aperture: addr[31:0] is the LDS byte offset
}

// ---------------------------------------------------------------------------
// Kernel 1: per-head projection  Y[bm] = X[b] @ W[m]^T   (f32 in, f16 out)
// Output layout: transpose==0 -> [t][i] ; transpose==1 -> [i][t]
// Grid: (D/64, T/64, B*M), 256 threads (8 waves), 64x64 output tile.
// ---------------------------------------------------------------------------
__global__ __launch_bounds__(256) void proj_kernel(
    const float* __restrict__ X, const float* __restrict__ W,
    _Float16* __restrict__ out, int transpose) {
  const int i0 = blockIdx.x * 64;
  const int t0 = blockIdx.y * 64;
  const int bm = blockIdx.z;
  const int b = bm / M_, m = bm % M_;
  const float* xb = X + (size_t)b * T_ * D_;
  const float* wm = W + (size_t)m * D_ * D_;
  _Float16* ob = out + (size_t)bm * T_ * D_;

  __shared__ _Float16 At[64 * 128];   // X tile  [t][k]
  __shared__ _Float16 Bt[64 * 128];   // W tile  [i][k]  (== B[n][k])

  const int tid = threadIdx.x;
  const int wave = tid >> 5, lane = tid & 31;
  const int r = wave >> 1, cw = wave & 1;
  const int half = lane >> 4, n = lane & 15;

  v8f acc[2] = {};

  for (int kc = 0; kc < 4; ++kc) {
    const int kb = kc * 128;
#pragma unroll
    for (int q = 0; q < 8; ++q) {           // 2048 float4 total
      int g = tid + q * 256;
      int row = g >> 5, c4 = g & 31;
      float4 va = *(const float4*)(xb + (size_t)(t0 + row) * D_ + kb + c4 * 4);
      _Float16* da = &At[row * 128 + c4 * 4];
      da[0] = (_Float16)va.x; da[1] = (_Float16)va.y;
      da[2] = (_Float16)va.z; da[3] = (_Float16)va.w;
      float4 vb = *(const float4*)(wm + (size_t)(i0 + row) * D_ + kb + c4 * 4);
      _Float16* db = &Bt[row * 128 + c4 * 4];
      db[0] = (_Float16)vb.x; db[1] = (_Float16)vb.y;
      db[2] = (_Float16)vb.z; db[3] = (_Float16)vb.w;
    }
    __syncthreads();
#pragma unroll
    for (int ks = 0; ks < 4; ++ks) {
      v16h a = ld_frag(At, 128, r * 16, ks * 32, lane);
#pragma unroll
      for (int ct = 0; ct < 2; ++ct) {
        v16h bf = ld_frag(Bt, 128, cw * 32 + ct * 16, ks * 32, lane);
        acc[ct] = wmma16(a, bf, acc[ct]);
      }
    }
    __syncthreads();
  }

#pragma unroll
  for (int ct = 0; ct < 2; ++ct)
#pragma unroll
    for (int j = 0; j < 8; ++j) {
      int tg = t0 + r * 16 + j + 8 * half;
      int ig = i0 + cw * 32 + ct * 16 + n;
      _Float16 v = (_Float16)acc[ct][j];
      if (transpose) ob[(size_t)ig * T_ + tg] = v;
      else           ob[(size_t)tg * D_ + ig] = v;
    }
}

// ---------------------------------------------------------------------------
// Kernel 2: causal flash attention for one (b,m) and a 16-query tile.
// QZ,KZ: [bm][t][i] f16 ; PZT: [bm][i][t] f16 ; OC: [b][t][m*D+i] f16.
// K/V tiles arrive via Tensor Data Mover (wave 0 issues, s_wait_tensorcnt).
// S = Q*K^T split across all 8 waves (k-quarters), reduced through LDS.
// Grid: (T/16, B*M), 256 threads.
// ---------------------------------------------------------------------------
__global__ __launch_bounds__(256) void attn_kernel(
    const _Float16* __restrict__ QZ, const _Float16* __restrict__ KZ,
    const _Float16* __restrict__ PZT, _Float16* __restrict__ OC) {
  const int q0 = blockIdx.x * 16;
  const int bm = blockIdx.y;
  const int b = bm / M_, m = bm % M_;
  const _Float16* Qz = QZ + (size_t)bm * T_ * D_;
  const _Float16* Kz = KZ + (size_t)bm * T_ * D_;
  const _Float16* Pz = PZT + (size_t)bm * T_ * D_;   // [i][t]

  __shared__ _Float16 Qt[16 * 512];    // queries [t][k]        16 KB
  __shared__ _Float16 Kt[32 * 512];    // keys    [s][k]        32 KB (TDM)
  __shared__ _Float16 Pt[512 * 32];    // values  [i][s]        32 KB (TDM)
  __shared__ _Float16 Sp[16 * 32];     // probs   [t][s]         1 KB
  __shared__ float Spart[8][256];      // partial S (f32)        8 KB
  __shared__ float cmax[2][16], csum[2][16];
  __shared__ float rowm[16], rowl[16], mnew[16], rsc[16];

  const int tid = threadIdx.x;
  const int wave = tid >> 5, lane = tid & 31;
  const int half = lane >> 4, n = lane & 15;
  const int kc = wave & 1;         // key chunk (16 keys) this wave reduces
  const int kq = wave >> 1;        // k-quarter (128 wide) this wave multiplies

  // Load Q tile (16x512 f16 = 1024 uint4).
#pragma unroll
  for (int q = 0; q < 4; ++q) {
    int g = tid + q * 256;
    int row = g >> 6, c = g & 63;
    *(uint4*)&Qt[row * 512 + c * 8] =
        *(const uint4*)(Qz + (size_t)(q0 + row) * D_ + c * 8);
  }
  if (tid < 16) { rowm[tid] = -3.0e38f; rowl[tid] = 0.0f; }

  const unsigned ldsK = lds_offset_of(&Kt[0]);
  const unsigned ldsP = lds_offset_of(&Pt[0]);

  v8f acc[4] = {};
  const int nkt = (q0 + 15) / 32 + 1;   // causal key-tile bound

  for (int it = 0; it < nkt; ++it) {
    const int s0 = it * 32;
    __syncthreads();   // previous-iteration Kt/Pt/Sp consumers done

    if (it + 1 < nkt)  // warm next K tile (global_prefetch_b8)
      __builtin_prefetch(Kz + (size_t)(s0 + 32) * D_, 0, 1);

    if (wave == 0) {
      // K tile: 32 rows x 512 elems, row pitch D_.
      tdm_load_2d(Kz + (size_t)s0 * D_, ldsK, D_, 32, D_, (unsigned)(T_ - s0), D_);
      // V tile: 512 rows x 32 elems from [i][t] tensor, row pitch T_.
      tdm_load_2d(Pz + s0, ldsP, 32, D_, (unsigned)(T_ - s0), D_, T_);
      __builtin_amdgcn_s_wait_tensorcnt(0);
    }
    __syncthreads();

    // Partial S: wave (kc,kq) computes 16 queries x 16 keys over 128-wide k.
    v8f sp = {};
#pragma unroll
    for (int ks = 0; ks < 4; ++ks) {
      v16h a  = ld_frag(Qt, 512, 0, kq * 128 + ks * 32, lane);
      v16h bf = ld_frag(Kt, 512, kc * 16, kq * 128 + ks * 32, lane);
      sp = wmma16(a, bf, sp);
    }
#pragma unroll
    for (int j = 0; j < 8; ++j)
      Spart[wave][(j + 8 * half) * 16 + n] = sp[j];
    __syncthreads();

    // Waves 0-1 reduce k-quarters, scale, mask, per-chunk row max.
    v8f sacc = {};
    if (wave < 2) {
      const float scl = 0.04419417382415922f;   // 1/sqrt(512)
#pragma unroll
      for (int j = 0; j < 8; ++j) {
        int idx = (j + 8 * half) * 16 + n;
        float v = (Spart[wave][idx] + Spart[wave + 2][idx] +
                   Spart[wave + 4][idx] + Spart[wave + 6][idx]) * scl;
        int tg = q0 + j + 8 * half;
        int sg = s0 + wave * 16 + n;
        if (sg > tg) v = -3.0e38f;
        sacc[j] = v;
        float r = v;
        r = fmaxf(r, __shfl_xor(r, 1, 32));
        r = fmaxf(r, __shfl_xor(r, 2, 32));
        r = fmaxf(r, __shfl_xor(r, 4, 32));
        r = fmaxf(r, __shfl_xor(r, 8, 32));
        if (n == 0) cmax[wave][j + 8 * half] = r;
      }
    }
    __syncthreads();
    if (tid < 16) {
      float mo = rowm[tid];
      float mn = fmaxf(mo, fmaxf(cmax[0][tid], cmax[1][tid]));
      mnew[tid] = mn;
      rsc[tid]  = __expf(mo - mn);
    }
    __syncthreads();
    // exp, probs to LDS (f16), per-chunk row sums.
    if (wave < 2) {
#pragma unroll
      for (int j = 0; j < 8; ++j) {
        int row = j + 8 * half;
        float p = __expf(sacc[j] - mnew[row]);
        Sp[row * 32 + wave * 16 + n] = (_Float16)p;
        float r = p;
        r += __shfl_xor(r, 1, 32);
        r += __shfl_xor(r, 2, 32);
        r += __shfl_xor(r, 4, 32);
        r += __shfl_xor(r, 8, 32);
        if (n == 0) csum[wave][row] = r;
      }
    }
    __syncthreads();
    if (tid < 16) {
      rowl[tid] = rowl[tid] * rsc[tid] + csum[0][tid] + csum[1][tid];
      rowm[tid] = mnew[tid];
    }
    __syncthreads();

    // Rescale running O and accumulate O += P(16x32) * PZ(32x64-per-wave).
    float f[8];
#pragma unroll
    for (int j = 0; j < 8; ++j) f[j] = rsc[j + 8 * half];
    v16h a = ld_frag(Sp, 32, 0, 0, lane);
#pragma unroll
    for (int ct = 0; ct < 4; ++ct) {
#pragma unroll
      for (int j = 0; j < 8; ++j) acc[ct][j] *= f[j];
      v16h bf = ld_frag(Pt, 32, wave * 64 + ct * 16, 0, lane);
      acc[ct] = wmma16(a, bf, acc[ct]);
    }
  }

  // Normalize and store concat layout [b][t][m*512 + i].
  float fl[8];
#pragma unroll
  for (int j = 0; j < 8; ++j) fl[j] = 1.0f / rowl[j + 8 * half];
#pragma unroll
  for (int ct = 0; ct < 4; ++ct)
#pragma unroll
    for (int j = 0; j < 8; ++j) {
      int tg = q0 + j + 8 * half;
      int col = m * D_ + wave * 64 + ct * 16 + n;
      OC[((size_t)(b * T_ + tg)) * (M_ * D_) + col] = (_Float16)(acc[ct][j] * fl[j]);
    }
}

// ---------------------------------------------------------------------------
// Kernel 3: output projection + bias + residual.
// H[t][d] = sum_c OC[t][c] * Wo[d][c] + bo[d] + X[t][d]   (c over 4096)
// A (f16) tile arrives via TDM; B (f32 Wo) staged+converted by VALU.
// Grid: (D/64, B*T/64), 256 threads.
// ---------------------------------------------------------------------------
__global__ __launch_bounds__(256) void outproj_kernel(
    const _Float16* __restrict__ OC, const float* __restrict__ Wo,
    const float* __restrict__ bo, const float* __restrict__ X,
    float* __restrict__ H) {
  const int d0 = blockIdx.x * 64;
  const int t0 = blockIdx.y * 64;
  __shared__ _Float16 At[64 * 128];   // OC tile [t][c]  (TDM)
  __shared__ _Float16 Bt[64 * 128];   // Wo tile [d][c]

  const int tid = threadIdx.x;
  const int wave = tid >> 5, lane = tid & 31;
  const int r = wave >> 1, cw = wave & 1;
  const int half = lane >> 4, n = lane & 15;
  const unsigned ldsA = lds_offset_of(&At[0]);

  v8f acc[2] = {};

  for (int kc = 0; kc < 32; ++kc) {
    const int kb = kc * 128;
    if (wave == 0) {
      tdm_load_2d(OC + (size_t)t0 * (M_ * D_) + kb, ldsA, 128, 64,
                  (unsigned)(M_ * D_ - kb), (unsigned)(B_ * T_ - t0), M_ * D_);
    }
#pragma unroll
    for (int q = 0; q < 8; ++q) {          // 2048 float4 (f32 B)
      int g = tid + q * 256;
      int row = g >> 5, c4 = g & 31;
      float4 v = *(const float4*)(Wo + (size_t)(d0 + row) * (M_ * D_) + kb + c4 * 4);
      _Float16* db = &Bt[row * 128 + c4 * 4];
      db[0] = (_Float16)v.x; db[1] = (_Float16)v.y;
      db[2] = (_Float16)v.z; db[3] = (_Float16)v.w;
    }
    if (wave == 0) __builtin_amdgcn_s_wait_tensorcnt(0);
    __syncthreads();
#pragma unroll
    for (int ks = 0; ks < 4; ++ks) {
      v16h a = ld_frag(At, 128, r * 16, ks * 32, lane);
#pragma unroll
      for (int ct = 0; ct < 2; ++ct) {
        v16h bf = ld_frag(Bt, 128, cw * 32 + ct * 16, ks * 32, lane);
        acc[ct] = wmma16(a, bf, acc[ct]);
      }
    }
    __syncthreads();
  }

#pragma unroll
  for (int ct = 0; ct < 2; ++ct)
#pragma unroll
    for (int j = 0; j < 8; ++j) {
      int tg = t0 + r * 16 + j + 8 * half;
      int dg = d0 + cw * 32 + ct * 16 + n;
      float h = acc[ct][j] + bo[dg] + X[(size_t)tg * D_ + dg];
      H[(size_t)tg * D_ + dg] = h;
    }
}

// ---------------------------------------------------------------------------
// Kernel 4: LayerNorm over rows of 512. Grid: B*T blocks x 256 threads.
// ---------------------------------------------------------------------------
__global__ __launch_bounds__(256) void ln_kernel(
    const float* __restrict__ H, const float* __restrict__ gamma,
    const float* __restrict__ beta, float* __restrict__ out) {
  const int row = blockIdx.x;
  const int tid = threadIdx.x;
  const float* hr = H + (size_t)row * D_;
  float v0 = hr[tid], v1 = hr[tid + 256];

  __shared__ float s1[256], s2[256];
  s1[tid] = v0 + v1;
  s2[tid] = v0 * v0 + v1 * v1;
  __syncthreads();
  for (int off = 128; off > 0; off >>= 1) {
    if (tid < off) { s1[tid] += s1[tid + off]; s2[tid] += s2[tid + off]; }
    __syncthreads();
  }
  float mu  = s1[0] * (1.0f / D_);
  float var = s2[0] * (1.0f / D_) - mu * mu;
  float rs  = rsqrtf(var + LN_EPS_);
  float* orow = out + (size_t)row * D_;
  orow[tid]       = (v0 - mu) * rs * gamma[tid]       + beta[tid];
  orow[tid + 256] = (v1 - mu) * rs * gamma[tid + 256] + beta[tid + 256];
}

// ---------------------------------------------------------------------------
extern "C" void kernel_launch(void* const* d_in, const int* in_sizes, int n_in,
                              void* d_out, int out_size, void* d_ws, size_t ws_size,
                              hipStream_t stream) {
  const float* x     = (const float*)d_in[0];
  const float* P     = (const float*)d_in[1];
  const float* Q     = (const float*)d_in[2];
  const float* K     = (const float*)d_in[3];
  const float* Wo    = (const float*)d_in[4];
  const float* bo    = (const float*)d_in[5];
  const float* gamma = (const float*)d_in[6];
  const float* beta  = (const float*)d_in[7];
  float* out = (float*)d_out;

  const size_t nBMTD = (size_t)B_ * M_ * T_ * D_;   // 33,554,432 elems
  _Float16* QZ  = (_Float16*)d_ws;                  // 64 MiB
  _Float16* KZ  = QZ + nBMTD;                       // 64 MiB
  _Float16* PZT = KZ + nBMTD;                       // 64 MiB  ([bm][i][t])
  _Float16* OC  = PZT + nBMTD;                      // 64 MiB  ([b][t][m*D+i])
  float*    H   = (float*)(OC + nBMTD);             // 16 MiB

  dim3 gproj(D_ / 64, T_ / 64, B_ * M_);
  proj_kernel<<<gproj, 256, 0, stream>>>(x, Q, QZ, 0);
  proj_kernel<<<gproj, 256, 0, stream>>>(x, K, KZ, 0);
  proj_kernel<<<gproj, 256, 0, stream>>>(x, P, PZT, 1);

  attn_kernel<<<dim3(T_ / 16, B_ * M_), 256, 0, stream>>>(QZ, KZ, PZT, OC);

  outproj_kernel<<<dim3(D_ / 64, (B_ * T_) / 64), 256, 0, stream>>>(OC, Wo, bo, x, H);

  ln_kernel<<<B_ * T_, 256, 0, stream>>>(H, gamma, beta, out);
}
